// TransformerCharEncoder_46420006535355
// MI455X (gfx1250) — compile-verified, matrix operand order checked by reading, not run on hardware
//
#include <hip/hip_runtime.h>
#include <hip/hip_bf16.h>

// ---------------- problem constants (match reference) ----------------
#define BB   8
#define SS   1024
#define INF_ 256     // IN
#define DD   512
#define HH   8
#define LL   2
#define MM   64
#define DK   64
#define FFD  2048
#define EPS  1e-5f

typedef unsigned short u16;
typedef unsigned int   u32;
typedef unsigned long long u64;

typedef __attribute__((ext_vector_type(16))) __bf16 v16bf;
typedef __attribute__((ext_vector_type(8)))  float  v8f;
typedef __attribute__((ext_vector_type(4)))  unsigned int u32x4;
typedef __attribute__((ext_vector_type(8)))  int i32x8;
typedef __attribute__((ext_vector_type(4)))  int i32x4;

union FragU { uint4 q[2]; v16bf v; };

// ---- Tensor Data Mover availability (device pass only) ----
#if defined(__has_builtin)
#if __has_builtin(__builtin_amdgcn_tensor_load_to_lds) && defined(__AMDGCN__)
#define HAVE_TDM 1
#endif
#endif
#ifndef HAVE_TDM
#define HAVE_TDM 0
#endif

#if HAVE_TDM
// Issue a 2D TDM load of bf16 (2-byte) elements: tile_d1 rows x tile_d0 elems,
// row stride stride0 (elems), from global gaddr into LDS byte offset lds_off.
__device__ __forceinline__ void tdm_load_bf16_2d(u32 lds_off, const void* gaddr,
                                                 u32 tile_d0, u32 tile_d1,
                                                 u32 tensor_d0, u32 tensor_d1,
                                                 u64 stride0)
{
    u64 ga = (u64)(uintptr_t)gaddr;
    u32x4 g0;
    g0[0] = 1u;                                           // count=1, user mode
    g0[1] = lds_off;                                      // lds_addr (bytes)
    g0[2] = (u32)ga;                                      // global_addr[31:0]
    g0[3] = ((u32)(ga >> 32) & 0x01FFFFFFu) | (2u << 30); // addr[56:32] | type=2
    i32x8 g1;
    g1[0] = (int)(1u << 16);                              // data_size=1 -> 2 bytes
    g1[1] = (int)((tensor_d0 & 0xFFFFu) << 16);           // tensor_dim0[15:0]
    g1[2] = (int)((tensor_d0 >> 16) | ((tensor_d1 & 0xFFFFu) << 16));
    g1[3] = (int)((tensor_d1 >> 16) | (tile_d0 << 16));   // tile_dim0
    g1[4] = (int)(tile_d1 & 0xFFFFu);                     // tile_dim1 (tile_dim2=0)
    g1[5] = (int)(u32)stride0;                            // tensor_dim0_stride[31:0]
    g1[6] = (int)((u32)(stride0 >> 32) & 0xFFFFu);        // stride0[47:32], stride1=0
    g1[7] = 0;
    i32x4 z4 = (i32x4)0;
#if __clang_major__ >= 23
    i32x8 z8 = (i32x8)0;
    __builtin_amdgcn_tensor_load_to_lds(g0, g1, z4, z4, z8, 0);
#else
    __builtin_amdgcn_tensor_load_to_lds(g0, g1, z4, z4, 0);
#endif
}
#define TDM_WAIT(n) __builtin_amdgcn_s_wait_tensorcnt((short)(n))
#endif

__device__ __forceinline__ u16 f2b(float f) {
    u32 u = __float_as_uint(f);
    u32 r = u + 0x7FFFu + ((u >> 16) & 1u);   // round-to-nearest-even
    return (u16)(r >> 16);
}

// ---------------- fp32 -> bf16 convert ----------------
__global__ __launch_bounds__(256) void cvt_bf16_kernel(const float* __restrict__ in,
                                                       u16* __restrict__ out, int n) {
    int i = blockIdx.x * 256 + threadIdx.x;
    if (i < n) out[i] = f2b(in[i]);
}

// ---------------- generic WMMA GEMM (double-buffered, TDM pipelined) ----------
// C[M x N] = epilogue( A_bf16[M x K] @ B_bf16[K x N] + bias )
// MODE bit0: relu, bit1: residual add (resid f32 [M x N]), bit2: add sinusoid PE
template<int MODE, bool OUTF, bool OUTB>
__global__ __launch_bounds__(256) void gemm_bf16_wmma(
    const u16* __restrict__ A, const u16* __restrict__ Bm,
    const float* __restrict__ bias, const float* __restrict__ resid,
    float* __restrict__ outF, u16* __restrict__ outB,
    int N, int K)
{
    __shared__ __align__(16) u16 As[2][128][32];
    __shared__ __align__(16) u16 Bs[2][64][32];

    const int tid  = threadIdx.x;
    const int lane = tid & 31;
    const int wave = tid >> 5;          // 8 waves: 4 (M) x 2 (N)
    const int wm = wave >> 1, wn = wave & 1;
    const int r0 = blockIdx.x * 128;
    const int n0 = blockIdx.y * 64;
    const int kh = lane >> 4;           // which half of wave
    const int lc = lane & 15;
    const int T  = K / 32;              // K tiles
    const int krow = tid >> 3, nn = (tid & 7) * 8;   // B staging coords

    v8f c[2][2] = {};

    // prologue: B(0) into regs, TDM A(0) into buffer 0
    uint4 breg = *(const uint4*)&Bm[(size_t)krow * N + n0 + nn];
#if HAVE_TDM
    if (wave == 0)
        tdm_load_bf16_2d((u32)(uintptr_t)&As[0][0][0],
                         &A[(size_t)r0 * K], 32u, 128u, (u32)K, 1u << 20, (u64)K);
#endif

    for (int i = 0; i < T; ++i) {
        const int buf = i & 1;
        const int k0  = i * 32;

        // commit staged B regs for tile i into Bs[buf] (transposed, K contiguous)
        {
            const u16* bsr = (const u16*)&breg;
            #pragma unroll
            for (int t = 0; t < 8; ++t) Bs[buf][nn + t][krow] = bsr[t];
        }
#if !HAVE_TDM
        // fallback: manual A tile staging into As[buf]
        #pragma unroll
        for (int u = 0; u < 2; ++u) {
            int idx = tid + u * 256;
            int row = idx >> 2, cc = (idx & 3) * 8;
            *(uint4*)&As[buf][row][cc] =
                *(const uint4*)&A[(size_t)(r0 + row) * K + k0 + cc];
        }
#endif
        __syncthreads();   // retires compute(i-1): safe to overwrite As[buf^1]

#if HAVE_TDM
        if (i + 1 < T && wave == 0)   // prefetch A(i+1) via tensor DMA
            tdm_load_bf16_2d((u32)(uintptr_t)&As[buf ^ 1][0][0],
                             &A[(size_t)r0 * K + k0 + 32],
                             32u, 128u, (u32)K, 1u << 20, (u64)K);
#endif
        if (i + 1 < T)                // prefetch B(i+1) into regs
            breg = *(const uint4*)&Bm[(size_t)(k0 + 32 + krow) * N + n0 + nn];
        if (i + 2 < T)
            __builtin_prefetch(&Bm[(size_t)(k0 + 64 + krow) * N + n0 + nn], 0, 2);
#if HAVE_TDM
        if (wave == 0) {              // A(i) landed; A(i+1) may stay in flight
            if (i + 1 < T) TDM_WAIT(1); else TDM_WAIT(0);
        }
#endif
        __syncthreads();

        FragU a[2], b[2];
        #pragma unroll
        for (int i2 = 0; i2 < 2; ++i2) {
            const u16* ap = &As[buf][wm * 32 + i2 * 16 + lc][0];
            a[i2].q[0] = *(const uint4*)(ap + 8 * kh);
            a[i2].q[1] = *(const uint4*)(ap + 16 + 8 * kh);
            const u16* bp = &Bs[buf][wn * 32 + i2 * 16 + lc][0];
            b[i2].q[0] = *(const uint4*)(bp + 8 * kh);
            b[i2].q[1] = *(const uint4*)(bp + 16 + 8 * kh);
        }
        #pragma unroll
        for (int i2 = 0; i2 < 2; ++i2)
            #pragma unroll
            for (int j2 = 0; j2 < 2; ++j2)
                c[i2][j2] = __builtin_amdgcn_wmma_f32_16x16x32_bf16(
                    false, a[i2].v, false, b[j2].v, (short)0, c[i2][j2], false, false);
    }

    // epilogue (compile-time specialized)
    #pragma unroll
    for (int i = 0; i < 2; ++i) {
        #pragma unroll
        for (int j = 0; j < 2; ++j) {
            int gcol = n0 + wn * 32 + j * 16 + lc;
            float bi = bias[gcol];
            #pragma unroll
            for (int r = 0; r < 8; ++r) {
                int grow = r0 + wm * 32 + i * 16 + r + 8 * kh;
                float vv = c[i][j][r] + bi;
                if (MODE & 4) {  // sinusoid positional encoding
                    int pos = grow & (SS - 1);
                    float ang = (float)pos *
                        __expf(-9.210340372f * (float)(gcol & ~1) / (float)DD);
                    vv += (gcol & 1) ? __cosf(ang) : __sinf(ang);
                }
                size_t idx = (size_t)grow * N + gcol;
                if (MODE & 2) vv += resid[idx];
                if (MODE & 1) vv = fmaxf(vv, 0.f);
                if (OUTF) outF[idx] = vv;
                if (OUTB) outB[idx] = f2b(vv);
            }
        }
    }
}

// ---------------- LayerNorm: one wave (32 lanes) per row of D=512 ----------------
__global__ __launch_bounds__(256) void layernorm_kernel(
    const float* __restrict__ x, const float* __restrict__ g,
    const float* __restrict__ be, u16* __restrict__ outB, float* __restrict__ outF)
{
    int row  = blockIdx.x * 8 + (threadIdx.x >> 5);
    int lane = threadIdx.x & 31;
    const float* xr = x + (size_t)row * DD;

    float vals[16];
    float s = 0.f;
    #pragma unroll
    for (int i = 0; i < 4; ++i) {
        float4 t = *(const float4*)&xr[i * 128 + lane * 4];
        vals[i * 4 + 0] = t.x; vals[i * 4 + 1] = t.y;
        vals[i * 4 + 2] = t.z; vals[i * 4 + 3] = t.w;
        s += t.x + t.y + t.z + t.w;
    }
    #pragma unroll
    for (int m = 16; m > 0; m >>= 1) s += __shfl_xor(s, m, 32);
    float mean = s * (1.f / (float)DD);

    float s2 = 0.f;
    #pragma unroll
    for (int i = 0; i < 16; ++i) { float d = vals[i] - mean; s2 += d * d; }
    #pragma unroll
    for (int m = 16; m > 0; m >>= 1) s2 += __shfl_xor(s2, m, 32);
    float rs = rsqrtf(s2 * (1.f / (float)DD) + EPS);

    #pragma unroll
    for (int i = 0; i < 4; ++i) {
        #pragma unroll
        for (int jj = 0; jj < 4; ++jj) {
            int col = i * 128 + lane * 4 + jj;
            float y = (vals[i * 4 + jj] - mean) * rs * g[col] + be[col];
            size_t idx = (size_t)row * DD + col;
            if (outB) outB[idx] = f2b(y);
            if (outF) outF[idx] = y;
        }
    }
}

// ---------------- flash-style attention with relative bias + key mask ----------------
// q,k,v,o: bf16 [B*S][D], head h occupies cols h*DK .. h*DK+63
__global__ __launch_bounds__(256) void attention_kernel(
    const u16* __restrict__ q, const u16* __restrict__ k, const u16* __restrict__ v,
    const float* __restrict__ rel_bias, const int* __restrict__ lengths,
    u16* __restrict__ o)
{
    __shared__ __align__(16) u16 Qs[64][64];
    __shared__ __align__(16) u16 Ks[2][64][64];  // double buffer for TDM pipeline
    __shared__ __align__(16) u16 Vt[64][64];     // transposed: [dk][key]
    __shared__ __align__(16) u16 Ps[64][64];     // probabilities (bf16)
    __shared__ float Ss[64][64];
    __shared__ float m_s[64], l_s[64], csc[64];
    __shared__ float rb_s[2 * MM - 1];

    const int qt = blockIdx.x, h = blockIdx.y, b = blockIdx.z;
    const int q0 = qt * 64;
    const int tid = threadIdx.x, lane = tid & 31, wave = tid >> 5;
    const int rq = wave >> 1, ch = wave & 1;   // 4 row-groups x 2 dk/col halves
    const int kh = lane >> 4, lc = lane & 15;
    const int len = lengths[b];
    const int NT = SS / 64;

    if (tid < 2 * MM - 1) rb_s[tid] = rel_bias[(size_t)h * (2 * MM - 1) + tid];
    if (tid < 64) { m_s[tid] = -1e30f; l_s[tid] = 0.f; }

#if HAVE_TDM
    if (wave == 0) {
        tdm_load_bf16_2d((u32)(uintptr_t)&Qs[0][0],
                         &q[((size_t)(b * SS + q0)) * DD + h * DK],
                         64u, 64u, 1u << 20, 1u << 20, (u64)DD);
        tdm_load_bf16_2d((u32)(uintptr_t)&Ks[0][0][0],
                         &k[((size_t)(b * SS)) * DD + h * DK],
                         64u, 64u, 1u << 20, 1u << 20, (u64)DD);
    }
#else
    #pragma unroll
    for (int u = 0; u < 2; ++u) {
        int idx = tid + u * 256;                 // 512 uint4 covers 64x64 u16
        int r = idx >> 3, c8 = (idx & 7) * 8;
        *(uint4*)&Qs[r][c8] =
            *(const uint4*)&q[((size_t)(b * SS + q0 + r)) * DD + h * DK + c8];
    }
#endif

    v8f acc[2] = {};

    for (int j = 0; j < NT; ++j) {
        const int kb = j * 64;
        const int jb = j & 1;
#if HAVE_TDM
        if (wave == 0 && j + 1 < NT)   // prefetch next K tile via tensor DMA
            tdm_load_bf16_2d((u32)(uintptr_t)&Ks[jb ^ 1][0][0],
                             &k[((size_t)(b * SS + kb + 64)) * DD + h * DK],
                             64u, 64u, 1u << 20, 1u << 20, (u64)DD);
#endif
        // stage transposed V tile (TDM cannot transpose)
        #pragma unroll
        for (int u = 0; u < 2; ++u) {
            int idx = tid + u * 256;
            int r = idx >> 3, c8 = (idx & 7) * 8;
#if !HAVE_TDM
            *(uint4*)&Ks[jb][r][c8] =
                *(const uint4*)&k[((size_t)(b * SS + kb + r)) * DD + h * DK + c8];
#endif
            uint4 vv = *(const uint4*)&v[((size_t)(b * SS + kb + r)) * DD + h * DK + c8];
            const u16* vp = (const u16*)&vv;
            #pragma unroll
            for (int i = 0; i < 8; ++i) Vt[c8 + i][r] = vp[i];
        }
#if HAVE_TDM
        if (wave == 0) {   // Q..K(j) landed; K(j+1) may stay in flight
            if (j + 1 < NT) TDM_WAIT(1); else TDM_WAIT(0);
        }
#endif
        __syncthreads();

        // scores S = scale * Q K^T + rel_bias, masked -> Ss
        #pragma unroll
        for (int t = 0; t < 2; ++t) {
            int n0 = ch * 32 + t * 16;
            v8f sc = {};
            #pragma unroll
            for (int ks = 0; ks < 2; ++ks) {
                FragU a, bfr;
                const u16* ap = &Qs[rq * 16 + lc][ks * 32];
                a.q[0] = *(const uint4*)(ap + 8 * kh);
                a.q[1] = *(const uint4*)(ap + 16 + 8 * kh);
                const u16* bp = &Ks[jb][n0 + lc][ks * 32];
                bfr.q[0] = *(const uint4*)(bp + 8 * kh);
                bfr.q[1] = *(const uint4*)(bp + 16 + 8 * kh);
                sc = __builtin_amdgcn_wmma_f32_16x16x32_bf16(
                    false, a.v, false, bfr.v, (short)0, sc, false, false);
            }
            int ncol = n0 + lc;
            int gk = kb + ncol;
            bool valid = gk < len;
            #pragma unroll
            for (int r = 0; r < 8; ++r) {
                int mrow = rq * 16 + r + 8 * kh;
                int gq = q0 + mrow;
                int rel = gq - gk;
                rel = rel < -(MM - 1) ? -(MM - 1) : (rel > (MM - 1) ? (MM - 1) : rel);
                float sv = sc[r] * 0.125f + rb_s[rel + (MM - 1)];
                Ss[mrow][ncol] = valid ? sv : -1e30f;
            }
        }
        __syncthreads();

        // online softmax update (one thread per query row)
        if (tid < 64) {
            float mo = m_s[tid], mn = mo;
            #pragma unroll 8
            for (int cidx = 0; cidx < 64; ++cidx) mn = fmaxf(mn, Ss[tid][cidx]);
            float cf = __expf(mo - mn);
            float sum = 0.f;
            #pragma unroll 8
            for (int cidx = 0; cidx < 64; ++cidx) {
                float p = __expf(Ss[tid][cidx] - mn);
                sum += p;
                Ps[tid][cidx] = f2b(p);
            }
            m_s[tid] = mn;
            l_s[tid] = l_s[tid] * cf + sum;
            csc[tid] = cf;
        }
        __syncthreads();

        // rescale running accumulator, then acc += P @ V
        #pragma unroll
        for (int t = 0; t < 2; ++t)
            #pragma unroll
            for (int r = 0; r < 8; ++r)
                acc[t][r] *= csc[rq * 16 + r + 8 * kh];

        #pragma unroll
        for (int t = 0; t < 2; ++t) {
            int d0 = ch * 32 + t * 16;
            #pragma unroll
            for (int ks = 0; ks < 2; ++ks) {
                FragU a, bfr;
                const u16* ap = &Ps[rq * 16 + lc][ks * 32];
                a.q[0] = *(const uint4*)(ap + 8 * kh);
                a.q[1] = *(const uint4*)(ap + 16 + 8 * kh);
                const u16* bp = &Vt[d0 + lc][ks * 32];
                bfr.q[0] = *(const uint4*)(bp + 8 * kh);
                bfr.q[1] = *(const uint4*)(bp + 16 + 8 * kh);
                acc[t] = __builtin_amdgcn_wmma_f32_16x16x32_bf16(
                    false, a.v, false, bfr.v, (short)0, acc[t], false, false);
            }
        }
        __syncthreads();
    }

    // final normalize and store (bf16, ready for the Wo GEMM)
    #pragma unroll
    for (int t = 0; t < 2; ++t) {
        #pragma unroll
        for (int r = 0; r < 8; ++r) {
            int mrow = rq * 16 + r + 8 * kh;
            float ov = acc[t][r] / l_s[mrow];
            o[((size_t)(b * SS + q0 + mrow)) * DD + h * DK + ch * 32 + t * 16 + lc] = f2b(ov);
        }
    }
}

// ---------------- host orchestration ----------------
static inline void cvt(const float* in, u16* out, int n, hipStream_t s) {
    cvt_bf16_kernel<<<(n + 255) / 256, 256, 0, s>>>(in, out, n);
}

extern "C" void kernel_launch(void* const* d_in, const int* in_sizes, int n_in,
                              void* d_out, int out_size, void* d_ws, size_t ws_size,
                              hipStream_t stream) {
    const float* inputs  = (const float*)d_in[0];
    const int*   lengths = (const int*)  d_in[1];
    const float* W_in = (const float*)d_in[2];  const float* b_in = (const float*)d_in[3];
    const float* Wq = (const float*)d_in[4];    const float* bq = (const float*)d_in[5];
    const float* Wk = (const float*)d_in[6];    const float* bk = (const float*)d_in[7];
    const float* Wv = (const float*)d_in[8];    const float* bv = (const float*)d_in[9];
    const float* Wo = (const float*)d_in[10];   const float* bo = (const float*)d_in[11];
    const float* rel_bias = (const float*)d_in[12];
    const float* W1 = (const float*)d_in[13];   const float* b1 = (const float*)d_in[14];
    const float* W2 = (const float*)d_in[15];   const float* b2 = (const float*)d_in[16];
    const float* g1 = (const float*)d_in[17];   const float* be1 = (const float*)d_in[18];
    const float* g2 = (const float*)d_in[19];   const float* be2 = (const float*)d_in[20];
    const float* gf = (const float*)d_in[21];   const float* bef = (const float*)d_in[22];

    const int ROWS = BB * SS;   // 8192

    // carve workspace
    char* ws = (char*)d_ws;
    auto carve = [&](size_t bytes) -> void* {
        void* p = (void*)ws;
        ws += (bytes + 255) & ~(size_t)255;
        return p;
    };
    u16* wInB = (u16*)carve((size_t)INF_ * DD * 2);
    u16* wqB  = (u16*)carve((size_t)LL * DD * DD * 2);
    u16* wkB  = (u16*)carve((size_t)LL * DD * DD * 2);
    u16* wvB  = (u16*)carve((size_t)LL * DD * DD * 2);
    u16* woB  = (u16*)carve((size_t)LL * DD * DD * 2);
    u16* w1B  = (u16*)carve((size_t)LL * DD * FFD * 2);
    u16* w2B  = (u16*)carve((size_t)LL * FFD * DD * 2);
    u16* inB  = (u16*)carve((size_t)ROWS * INF_ * 2);
    float* x  = (float*)carve((size_t)ROWS * DD * 4);
    u16* hB   = (u16*)carve((size_t)ROWS * DD * 2);
    u16* qB   = (u16*)carve((size_t)ROWS * DD * 2);
    u16* kB   = (u16*)carve((size_t)ROWS * DD * 2);
    u16* vB   = (u16*)carve((size_t)ROWS * DD * 2);
    u16* oB   = (u16*)carve((size_t)ROWS * DD * 2);
    u16* ffB  = (u16*)carve((size_t)ROWS * FFD * 2);

    // bf16 conversions
    cvt(W_in, wInB, INF_ * DD, stream);
    cvt(Wq, wqB, LL * DD * DD, stream);
    cvt(Wk, wkB, LL * DD * DD, stream);
    cvt(Wv, wvB, LL * DD * DD, stream);
    cvt(Wo, woB, LL * DD * DD, stream);
    cvt(W1, w1B, LL * DD * FFD, stream);
    cvt(W2, w2B, LL * FFD * DD, stream);
    cvt(inputs, inB, ROWS * INF_, stream);

    dim3 blk(256);
    dim3 gD(ROWS / 128, DD / 64);     // N = 512
    dim3 gF(ROWS / 128, FFD / 64);    // N = 2048
    dim3 gLN(ROWS / 8);
    dim3 gAT(SS / 64, HH, BB);

    // embed: x = inputs @ W_in + b_in + PE     (MODE 4 = add PE, f32 out)
    gemm_bf16_wmma<4, true, false><<<gD, blk, 0, stream>>>(
        inB, wInB, b_in, nullptr, x, nullptr, DD, INF_);

    for (int l = 0; l < LL; ++l) {
        const u16* wq_l = wqB + (size_t)l * DD * DD;
        const u16* wk_l = wkB + (size_t)l * DD * DD;
        const u16* wv_l = wvB + (size_t)l * DD * DD;
        const u16* wo_l = woB + (size_t)l * DD * DD;
        const u16* w1_l = w1B + (size_t)l * DD * FFD;
        const u16* w2_l = w2B + (size_t)l * FFD * DD;
        const float* bq_l = bq + l * DD;  const float* bk_l = bk + l * DD;
        const float* bv_l = bv + l * DD;  const float* bo_l = bo + l * DD;
        const float* b1_l = b1 + l * FFD; const float* b2_l = b2 + l * DD;
        const float* g1_l = g1 + l * DD;  const float* be1_l = be1 + l * DD;
        const float* g2_l = g2 + l * DD;  const float* be2_l = be2 + l * DD;
        const float* rb_l = rel_bias + (size_t)l * HH * (2 * MM - 1);

        // pre-norm 1
        layernorm_kernel<<<gLN, blk, 0, stream>>>(x, g1_l, be1_l, hB, nullptr);
        // Q,K,V projections (bf16 outputs)
        gemm_bf16_wmma<0, false, true><<<gD, blk, 0, stream>>>(
            hB, wq_l, bq_l, nullptr, nullptr, qB, DD, DD);
        gemm_bf16_wmma<0, false, true><<<gD, blk, 0, stream>>>(
            hB, wk_l, bk_l, nullptr, nullptr, kB, DD, DD);
        gemm_bf16_wmma<0, false, true><<<gD, blk, 0, stream>>>(
            hB, wv_l, bv_l, nullptr, nullptr, vB, DD, DD);
        // attention
        attention_kernel<<<gAT, blk, 0, stream>>>(qB, kB, vB, rb_l, lengths, oB);
        // output projection + residual (MODE 2, f32 out)
        gemm_bf16_wmma<2, true, false><<<gD, blk, 0, stream>>>(
            oB, wo_l, bo_l, x, x, nullptr, DD, DD);
        // pre-norm 2
        layernorm_kernel<<<gLN, blk, 0, stream>>>(x, g2_l, be2_l, hB, nullptr);
        // FFN1 with ReLU (MODE 1), bf16 out
        gemm_bf16_wmma<1, false, true><<<gF, blk, 0, stream>>>(
            hB, w1_l, b1_l, nullptr, nullptr, ffB, FFD, DD);
        // FFN2 + residual (MODE 2, f32 out)
        gemm_bf16_wmma<2, true, false><<<gD, blk, 0, stream>>>(
            ffB, w2_l, b2_l, x, x, nullptr, DD, FFD);
    }

    // final LayerNorm -> fp32 output
    layernorm_kernel<<<gLN, blk, 0, stream>>>(x, gf, bef, nullptr, (float*)d_out);
}
